// DTMLayer_39556648796334
// MI455X (gfx1250) — compile-verified, matrix operand order checked by reading, not run on hardware
//
#include <hip/hip_runtime.h>
#include <hip/hip_bf16.h>

typedef __attribute__((ext_vector_type(2)))  float    v2f;
typedef __attribute__((ext_vector_type(8)))  float    v8f;
typedef __attribute__((ext_vector_type(16))) _Float16 v16h;

constexpr int   kB  = 32;
constexpr int   kN  = 1024;
constexpr float kM0 = 0.05f;
constexpr float kINF = 3.0e38f;

// One workgroup = 16 waves (512 threads) handles one batch b and one tile of
// 16 query rows.
// Phase 1: 16x1024 squared-distance tile via V_WMMA_F32_16X16X4_F32 (gram
//          trick, K=2 zero-padded to 4), written to a 64KB LDS tile.
// Phase 2: one wave per row; 32 (d2,w) candidates per lane held in registers;
//          deterministic cross-lane extract-min until weight mass reaches wb.
__global__ __launch_bounds__(512) void dtm_kernel(
    const float* __restrict__ inp,   // [B,N,2]
    const float* __restrict__ wgt,   // [B,N]
    const float* __restrict__ gpts,  // [N,2]
    float* __restrict__ out)         // [B,N]
{
    __shared__ float s_d2[16 * kN];          // 64 KB exactly

    const int tilesPerBatch = kN / 16;       // 64
    const int b     = blockIdx.x / tilesPerBatch;
    const int row0  = (blockIdx.x % tilesPerBatch) * 16;

    const int tid   = threadIdx.x;
    const int lane  = tid & 31;
    const int wid   = tid >> 5;              // 0..15
    const int idx16 = lane & 15;
    const int half  = lane >> 4;             // 0 = lanes 0-15, 1 = lanes 16-31
    const int mbase = half * 8;

    // ---------------- Phase 1: distance tile via WMMA ----------------
    // A-matrix row (query point) for this lane's slot: M = lane & 15.
    const float ax0 = inp[(b * kN + row0 + idx16) * 2 + 0];
    const float ax1 = inp[(b * kN + row0 + idx16) * 2 + 1];
    const float x2r = ax0 * ax0 + ax1 * ax1;  // |x|^2 of row (lane&15)

    // Each wave produces 4 of the 64 column tiles.
    for (int t = 0; t < 4; ++t) {
        const int c0 = (wid * 4 + t) * 16;
        const float g0 = gpts[(c0 + idx16) * 2 + 0];
        const float g1 = gpts[(c0 + idx16) * 2 + 1];
        const float g2 = g0 * g0 + g1 * g1;   // |g|^2 of column (lane&15)

        v8f acc = {0.f, 0.f, 0.f, 0.f, 0.f, 0.f, 0.f, 0.f};
#if __has_builtin(__builtin_amdgcn_wmma_f32_16x16x4_f32)
        // 16x4 f32 A: VGPR0 = K0 (lo half) / K2 (hi half), VGPR1 = K1 / K3.
        // K=2,3 are the zero padding -> upper half lanes carry zeros.
        v2f Av = { half ? 0.f : ax0, half ? 0.f : ax1 };
        v2f Bv = { half ? 0.f : g0,  half ? 0.f : g1  };
        acc = __builtin_amdgcn_wmma_f32_16x16x4_f32(
            false, Av, false, Bv, (short)0, acc, false, false);
#else
        // Fallback: codegen-confirmed f16 WMMA, K=0,1 populated, rest zero.
        v16h Ah, Bh;
        #pragma unroll
        for (int q = 0; q < 16; ++q) { Ah[q] = (_Float16)0.f; Bh[q] = (_Float16)0.f; }
        if (!half) {
            Ah[0] = (_Float16)ax0; Ah[1] = (_Float16)ax1;
            Bh[0] = (_Float16)g0;  Bh[1] = (_Float16)g1;
        }
        acc = __builtin_amdgcn_wmma_f32_16x16x32_f16(
            false, Ah, false, Bh, (short)0, acc, false, false);
#endif
        // D layout: VGPR i holds rows (i, i+8); column = lane & 15.
        #pragma unroll
        for (int i = 0; i < 8; ++i) {
            const float xx2 = __shfl(x2r, mbase + i, 32);
            float d2 = xx2 + g2 - 2.0f * acc[i];
            d2 = fmaxf(d2, 0.0f);             // == knn_dist^2 (R == 2)
            s_d2[(mbase + i) * kN + c0 + idx16] = d2;
        }
    }
    __syncthreads();

    // ---------------- Phase 2: weighted quantile per row ----------------
    float dd[32], ww[32];
    const float* drow = &s_d2[wid * kN];
    float wsum = 0.0f;
    #pragma unroll
    for (int j = 0; j < 32; ++j) {
        const int m = lane + 32 * j;
        dd[j] = drow[m];                      // conflict-free LDS reads
        ww[j] = wgt[b * kN + m];              // coalesced, L2-resident
        wsum += ww[j];
    }
    // wb = M0 * sum(weight[b,:]) via fixed-order wave reduction (deterministic)
    #pragma unroll
    for (int off = 16; off > 0; off >>= 1) wsum += __shfl_xor(wsum, off, 32);
    const float wb = kM0 * wsum;

    // Lane-local minimum (fully unrolled -> stays in VGPRs, no scratch)
    float lmin = kINF, lw = 0.0f;
    #pragma unroll
    for (int j = 0; j < 32; ++j)
        if (dd[j] < lmin) { lmin = dd[j]; lw = ww[j]; }

    float cum_w = 0.0f, cum_d = 0.0f, val = 0.0f;
    for (int it = 0; it < kN; ++it) {
        // Cross-lane argmin with deterministic (value, lane) tie-break.
        float mval = lmin; int msrc = lane; float mw = lw;
        #pragma unroll
        for (int off = 16; off > 0; off >>= 1) {
            const float ov = __shfl_xor(mval, off, 32);
            const int   os = __shfl_xor(msrc, off, 32);
            const float ow = __shfl_xor(mw,   off, 32);
            if (ov < mval || (ov == mval && os < msrc)) {
                mval = ov; msrc = os; mw = ow;
            }
        }
        if (cum_w + mw >= wb) {               // boundary neighbor reached
            val = cum_d + mval * (wb - cum_w);
            break;
        }
        cum_d += mval * mw;
        cum_w += mw;
        if (lane == msrc) {                   // winner retires its element
            bool removed = false;
            #pragma unroll
            for (int j = 0; j < 32; ++j)
                if (!removed && dd[j] == mval) { dd[j] = kINF; removed = true; }
            lmin = kINF; lw = 0.0f;
            #pragma unroll
            for (int j = 0; j < 32; ++j)
                if (dd[j] < lmin) { lmin = dd[j]; lw = ww[j]; }
        }
    }

    if (lane == 0) out[b * kN + row0 + wid] = sqrtf(val / wb);
}

extern "C" void kernel_launch(void* const* d_in, const int* in_sizes, int n_in,
                              void* d_out, int out_size, void* d_ws, size_t ws_size,
                              hipStream_t stream) {
    (void)in_sizes; (void)n_in; (void)out_size; (void)d_ws; (void)ws_size;
    const float* inp  = (const float*)d_in[0];   // [B,N,2] f32
    const float* wgt  = (const float*)d_in[1];   // [B,N]   f32
    const float* gpts = (const float*)d_in[2];   // [N,2]   f32
    float* out = (float*)d_out;                  // [B,N]   f32

    const int blocks = kB * (kN / 16);           // 32 * 64 = 2048 workgroups
    dtm_kernel<<<blocks, 512, 0, stream>>>(inp, wgt, gpts, out);
}
